// Block_15083925144394
// MI455X (gfx1250) — compile-verified, matrix operand order checked by reading, not run on hardware
//
#include <hip/hip_runtime.h>

// ---------------------------------------------------------------------------
// RWKV v4 block for MI455X (gfx1250, wave32, WMMA).
// GEMMs in bf16 with f32 accumulation via v_wmma_f32_16x16x32_bf16.
// Global->LDS staging uses CDNA5 async loads (ASYNCcnt) when ASYNC_STAGE=1.
// ---------------------------------------------------------------------------

#define ASYNC_STAGE 1

typedef __attribute__((ext_vector_type(8)))  __bf16 v8bf;
typedef __attribute__((ext_vector_type(16))) __bf16 v16bf;
typedef __attribute__((ext_vector_type(8)))  float  v8f;

#define BM 128
#define BN 128
#define BK 32

__device__ inline v16bf cat8(v8bf lo, v8bf hi) {
  return __builtin_shufflevector(lo, hi, 0,1,2,3,4,5,6,7,8,9,10,11,12,13,14,15);
}

__device__ inline float sigmoid_f(float x) { return 1.0f / (1.0f + __expf(-x)); }

#if ASYNC_STAGE
// 16B async copy: global memory -> LDS, tracked by ASYNCcnt (no VGPR data).
// VDST = LDS byte address (low 32 bits of the flat shared pointer),
// VADDR = 64-bit global address, SADDR = off.
__device__ __forceinline__ void async_ld16(unsigned int lds_addr, const void* gptr) {
  asm volatile("global_load_async_to_lds_b128 %0, %1, off"
               :: "v"(lds_addr), "v"(gptr)
               : "memory");
}
__device__ __forceinline__ void wait_async_le4() {
  asm volatile("s_wait_asynccnt 0x4" ::: "memory");
}
__device__ __forceinline__ void wait_async_0() {
  asm volatile("s_wait_asynccnt 0x0" ::: "memory");
}
__device__ __forceinline__ unsigned int lds_addr32(const void* p) {
  return (unsigned int)(unsigned long long)p;   // flat LDS addr: low 32 bits
}
#endif

// ---------------------------------------------------------------------------
// NT GEMM: C[m][n] = sum_k A[m][k] * W[n][k]
//   A: [M][K] bf16 row-major, W: [N][K] bf16 row-major (reference weights
//   are already [out][in] row-major, so no transpose anywhere).
// MODE: 0 = f32 store, 1 = sigmoid f32, 2 = relu^2 bf16,
//       3 = res + acc (f32), 4 = res + sigmoid(acc)*gate (f32)
// Requires M%128==0, N%128==0, K%32==0 (true for all shapes here).
// ---------------------------------------------------------------------------
template<int MODE>
__global__ __launch_bounds__(256)
void gemm_bf16_wmma(const __bf16* __restrict__ A, const __bf16* __restrict__ W,
                    float* __restrict__ outF, __bf16* __restrict__ outB,
                    const float* __restrict__ res, const float* __restrict__ gate,
                    int M, int N, int K)
{
  __shared__ __bf16 As[2][BM * BK];   // 16 KB
  __shared__ __bf16 Ws[2][BN * BK];   // 16 KB

  const int tid   = threadIdx.x;
  const int wave  = tid >> 5;
  const int lane  = tid & 31;
  const int waveM = wave >> 2;        // 0..1  -> 64 rows each
  const int waveN = wave & 3;         // 0..3  -> 32 cols each
  const int blockM = blockIdx.y * BM;
  const int blockN = blockIdx.x * BN;
  const int lh = lane & 15;           // row/col within 16-wide fragment
  const int kh = lane >> 4;           // K-half selector (per ISA A/B layout)

  // Staging: 512 chunks of 8 bf16 per tile; thread handles chunk tid & tid+256.
  const int rs = tid >> 2;            // 0..63
  const int cs = (tid & 3) * 8;       // 0,8,16,24
  const int aRow = blockM + rs;
  const int wRow = blockN + rs;

  v8f acc[4][2] = {};
  const int numK = K / BK;

#if ASYNC_STAGE
  const unsigned int laBase = lds_addr32(&As[0][rs * BK + cs]);
  const unsigned int lwBase = lds_addr32(&Ws[0][rs * BK + cs]);
  const unsigned int bufStride = (unsigned int)(BM * BK * 2);   // bytes per buffer
  const unsigned int rowHalf   = (unsigned int)(64 * BK * 2);   // +64 rows in bytes

  { // prologue: tile 0 -> LDS[0]
    const __bf16* ga = A + aRow * K + cs;
    const __bf16* gw = W + wRow * K + cs;
    async_ld16(laBase,            ga);
    async_ld16(laBase + rowHalf,  ga + 64 * K);
    async_ld16(lwBase,            gw);
    async_ld16(lwBase + rowHalf,  gw + 64 * K);
  }

  for (int kt = 0; kt < numK; ++kt) {
    const int buf = kt & 1;
    const bool hasNext = (kt + 1) < numK;
    if (hasNext) {  // issue next K-tile into the other buffer
      const int k0 = (kt + 1) * BK;
      const unsigned int nb = (unsigned int)(buf ^ 1) * bufStride;
      const __bf16* ga = A + aRow * K + k0 + cs;
      const __bf16* gw = W + wRow * K + k0 + cs;
      async_ld16(laBase + nb,           ga);
      async_ld16(laBase + nb + rowHalf, ga + 64 * K);
      async_ld16(lwBase + nb,           gw);
      async_ld16(lwBase + nb + rowHalf, gw + 64 * K);
      wait_async_le4();   // tile kt (issued earlier, in-order) has landed
    } else {
      wait_async_0();
    }
    __syncthreads();      // all waves' async arrivals for tile kt visible

    // fragments from LDS (layout per CDNA5 ISA 16-bit A 16x32)
    v16bf af[4];
#pragma unroll
    for (int mi = 0; mi < 4; ++mi) {
      const int r = waveM * 64 + mi * 16 + lh;
      const __bf16* p = &As[buf][r * BK + kh * 8];
      af[mi] = cat8(*(const v8bf*)p, *(const v8bf*)(p + 16));
    }
    v16bf bfr[2];
#pragma unroll
    for (int ni = 0; ni < 2; ++ni) {
      const int n = waveN * 32 + ni * 16 + lh;
      const __bf16* p = &Ws[buf][n * BK + kh * 8];
      bfr[ni] = cat8(*(const v8bf*)p, *(const v8bf*)(p + 16));
    }
#pragma unroll
    for (int mi = 0; mi < 4; ++mi)
#pragma unroll
      for (int ni = 0; ni < 2; ++ni)
        acc[mi][ni] = __builtin_amdgcn_wmma_f32_16x16x32_bf16(
            false, af[mi], false, bfr[ni], (short)0, acc[mi][ni], false, false);

    __syncthreads();      // readers done before buf is overwritten next iter
  }
#else
  { // prologue: tile 0 -> LDS[0] (register-staged fallback)
    const __bf16* ga = A + aRow * K + cs;
    const __bf16* gw = W + wRow * K + cs;
    v8bf a0 = *(const v8bf*)ga;
    v8bf a1 = *(const v8bf*)(ga + 64 * K);
    v8bf w0 = *(const v8bf*)gw;
    v8bf w1 = *(const v8bf*)(gw + 64 * K);
    *(v8bf*)&As[0][rs * BK + cs]        = a0;
    *(v8bf*)&As[0][(rs + 64) * BK + cs] = a1;
    *(v8bf*)&Ws[0][rs * BK + cs]        = w0;
    *(v8bf*)&Ws[0][(rs + 64) * BK + cs] = w1;
  }
  __syncthreads();
  for (int kt = 0; kt < numK; ++kt) {
    const int buf = kt & 1;
    const bool hasNext = (kt + 1) < numK;
    v8bf a0, a1, w0, w1;
    if (hasNext) {
      const int k0 = (kt + 1) * BK;
      const __bf16* ga = A + aRow * K + k0 + cs;
      const __bf16* gw = W + wRow * K + k0 + cs;
      a0 = *(const v8bf*)ga;
      a1 = *(const v8bf*)(ga + 64 * K);
      w0 = *(const v8bf*)gw;
      w1 = *(const v8bf*)(gw + 64 * K);
    }
    v16bf af[4];
#pragma unroll
    for (int mi = 0; mi < 4; ++mi) {
      const int r = waveM * 64 + mi * 16 + lh;
      const __bf16* p = &As[buf][r * BK + kh * 8];
      af[mi] = cat8(*(const v8bf*)p, *(const v8bf*)(p + 16));
    }
    v16bf bfr[2];
#pragma unroll
    for (int ni = 0; ni < 2; ++ni) {
      const int n = waveN * 32 + ni * 16 + lh;
      const __bf16* p = &Ws[buf][n * BK + kh * 8];
      bfr[ni] = cat8(*(const v8bf*)p, *(const v8bf*)(p + 16));
    }
#pragma unroll
    for (int mi = 0; mi < 4; ++mi)
#pragma unroll
      for (int ni = 0; ni < 2; ++ni)
        acc[mi][ni] = __builtin_amdgcn_wmma_f32_16x16x32_bf16(
            false, af[mi], false, bfr[ni], (short)0, acc[mi][ni], false, false);
    __syncthreads();
    if (hasNext) {
      const int nb = buf ^ 1;
      *(v8bf*)&As[nb][rs * BK + cs]        = a0;
      *(v8bf*)&As[nb][(rs + 64) * BK + cs] = a1;
      *(v8bf*)&Ws[nb][rs * BK + cs]        = w0;
      *(v8bf*)&Ws[nb][(rs + 64) * BK + cs] = w1;
    }
    __syncthreads();
  }
#endif

  // epilogue: C/D layout -> VGPR j holds (m = j + 8*kh, n = lh)
  // 32-bit indexing (max idx = 25600*2048 < 2^31) -> base + imm-offset stores.
  const int rBase = blockM + waveM * 64 + kh * 8;
  const int cBase = blockN + waveN * 32 + lh;
#pragma unroll
  for (int mi = 0; mi < 4; ++mi) {
#pragma unroll
    for (int ni = 0; ni < 2; ++ni) {
      const int base = (rBase + mi * 16) * N + cBase + ni * 16;
#pragma unroll
      for (int j = 0; j < 8; ++j) {
        const int idx = base + j * N;
        const float v = acc[mi][ni][j];
        if (MODE == 0)      outF[idx] = v;
        else if (MODE == 1) outF[idx] = sigmoid_f(v);
        else if (MODE == 2) { float t = fmaxf(v, 0.0f); outB[idx] = (__bf16)(t * t); }
        else if (MODE == 3) outF[idx] = res[idx] + v;
        else                outF[idx] = res[idx] + sigmoid_f(v) * gate[idx];
      }
    }
  }
}

// ---------------------------------------------------------------------------
// Fused LayerNorm + time-shift + lerp-mix. One block per token; computes
// LN statistics of token t AND t-1 in one pass, emits bf16 mixed streams.
// ---------------------------------------------------------------------------
__global__ __launch_bounds__(256)
void ln_mix_kernel(const float* __restrict__ x,
                   const float* __restrict__ g, const float* __restrict__ b,
                   const float* __restrict__ mk, const float* __restrict__ mv,
                   const float* __restrict__ mr,
                   __bf16* __restrict__ ok, __bf16* __restrict__ ov,
                   __bf16* __restrict__ orr,
                   int T, int C)
{
  const int tokenId = blockIdx.x;          // b*T + t
  const int t = tokenId % T;
  const int tid = threadIdx.x;
  const float* xt = x + tokenId * C;
  const float* xp = xt - C;
  const int c0 = tid, c1 = tid + 256;

  const float a0 = xt[c0], a1 = xt[c1];
  float p0 = 0.0f, p1 = 0.0f;
  if (t > 0) { p0 = xp[c0]; p1 = xp[c1]; }

  __shared__ float4 red[256];
  float4 s;
  s.x = a0 + a1;       s.y = a0 * a0 + a1 * a1;
  s.z = p0 + p1;       s.w = p0 * p0 + p1 * p1;
  red[tid] = s;
  __syncthreads();
  for (int off = 128; off > 0; off >>= 1) {
    if (tid < off) {
      float4 o = red[tid + off];
      red[tid].x += o.x; red[tid].y += o.y;
      red[tid].z += o.z; red[tid].w += o.w;
    }
    __syncthreads();
  }
  const float4 tot = red[0];
  const float invC = 1.0f / (float)C;
  const float mu   = tot.x * invC;
  const float rs   = rsqrtf(tot.y * invC - mu * mu + 1e-5f);
  const float mup  = tot.z * invC;
  const float rsp  = rsqrtf(tot.w * invC - mup * mup + 1e-5f);

#pragma unroll
  for (int e = 0; e < 2; ++e) {
    const int   c  = e ? c1 : c0;
    const float xv = e ? a1 : a0;
    const float pv = e ? p1 : p0;
    const float h  = (xv - mu) * rs * g[c] + b[c];
    const float hh = (t > 0) ? ((pv - mup) * rsp * g[c] + b[c]) : 0.0f;
    const int o = tokenId * C + c;
    const float wk = mk[c]; ok[o] = (__bf16)(h * wk + hh * (1.0f - wk));
    const float wv = mv[c]; ov[o] = (__bf16)(h * wv + hh * (1.0f - wv));
    if (orr) { const float wr = mr[c]; orr[o] = (__bf16)(h * wr + hh * (1.0f - wr)); }
  }
}

// ---------------------------------------------------------------------------
// WKV scan (numerically stable), fused with sigmoid(r) gate.
// One thread per (b, c); sequential over T=50; coalesced across c.
// ---------------------------------------------------------------------------
__global__ __launch_bounds__(256)
void wkv_kernel(const float* __restrict__ k, const float* __restrict__ v,
                const float* __restrict__ r,
                const float* __restrict__ dec, const float* __restrict__ fir,
                __bf16* __restrict__ y, int T, int C)
{
  const int idx = blockIdx.x * 256 + threadIdx.x;   // b*C + c
  const int c = idx % C;
  const float w = -__expf(dec[c]);
  const float u = fir[c];
  float aa = 0.0f, bb = 0.0f, pp = -1e38f;
  const int base = (idx / C) * T * C + c;
  for (int t = 0; t < T; ++t) {
    const int o = base + t * C;
    const float kt = k[o], vt = v[o];
    const float ww = u + kt;
    const float p  = fmaxf(pp, ww);
    const float e1 = __expf(pp - p);
    const float e2 = __expf(ww - p);
    const float yt = (e1 * aa + e2 * vt) / (e1 * bb + e2);
    y[o] = (__bf16)(r[o] * yt);
    const float ww2 = pp + w;
    const float p2  = fmaxf(ww2, kt);
    const float e1b = __expf(ww2 - p2);
    const float e2b = __expf(kt - p2);
    aa = e1b * aa + e2b * vt;
    bb = e1b * bb + e2b;
    pp = p2;
  }
}

// ---------------------------------------------------------------------------
// fp32 -> bf16 weight conversion (once per launch)
// ---------------------------------------------------------------------------
__global__ __launch_bounds__(256)
void cvt_f32_bf16(const float* __restrict__ s, __bf16* __restrict__ d, int n)
{
  const int i = blockIdx.x * 256 + threadIdx.x;
  if (i < n) d[i] = (__bf16)s[i];
}

// ---------------------------------------------------------------------------
extern "C" void kernel_launch(void* const* d_in, const int* in_sizes, int n_in,
                              void* d_out, int out_size, void* d_ws, size_t ws_size,
                              hipStream_t stream)
{
  (void)in_sizes; (void)n_in; (void)out_size; (void)ws_size;
  const int B = 512, T = 50, C = 512, H = 2048;
  const int M = B * T;                      // 25600 tokens

  const float* x     = (const float*)d_in[0];
  const float* ln1_g = (const float*)d_in[1];
  const float* ln1_b = (const float*)d_in[2];
  const float* ln2_g = (const float*)d_in[3];
  const float* ln2_b = (const float*)d_in[4];
  const float* amk   = (const float*)d_in[5];
  const float* amv   = (const float*)d_in[6];
  const float* amr   = (const float*)d_in[7];
  const float* tdec  = (const float*)d_in[8];
  const float* tfir  = (const float*)d_in[9];
  const float* Wk    = (const float*)d_in[10];
  const float* Wv    = (const float*)d_in[11];
  const float* Wr    = (const float*)d_in[12];
  const float* Wo    = (const float*)d_in[13];
  const float* fmk   = (const float*)d_in[14];
  const float* fmr   = (const float*)d_in[15];
  const float* Fk    = (const float*)d_in[16];
  const float* Fv    = (const float*)d_in[17];
  const float* Fr    = (const float*)d_in[18];

  float* out = (float*)d_out;

  // ---- workspace layout (slot-reused; ~243 MB) ----
  char* ws = (char*)d_ws;
  size_t off = 0;
  auto alloc = [&](size_t bytes) -> char* {
    char* p = ws + off;
    off += (bytes + 255) & ~(size_t)255;
    return p;
  };
  __bf16* xk  = (__bf16*)alloc((size_t)M * C * 2);
  __bf16* xv  = (__bf16*)alloc((size_t)M * C * 2);
  __bf16* xr  = (__bf16*)alloc((size_t)M * C * 2);   // later reused as y
  float*  kb  = (float*) alloc((size_t)M * C * 4);   // kb+vb later = kk (bf16 M*H)
  float*  vb  = (float*) alloc((size_t)M * C * 4);
  float*  rb  = (float*) alloc((size_t)M * C * 4);   // later reused as kv
  __bf16* wkb = (__bf16*)alloc((size_t)C * C * 2);
  __bf16* wvb = (__bf16*)alloc((size_t)C * C * 2);
  __bf16* wrb = (__bf16*)alloc((size_t)C * C * 2);
  __bf16* wob = (__bf16*)alloc((size_t)C * C * 2);
  __bf16* fkb = (__bf16*)alloc((size_t)H * C * 2);
  __bf16* fvb = (__bf16*)alloc((size_t)C * H * 2);
  __bf16* frb = (__bf16*)alloc((size_t)C * C * 2);

  // ---- weight conversion (small, once per launch) ----
  const int cc = C * C, hc = H * C;
  cvt_f32_bf16<<<(cc + 255) / 256, 256, 0, stream>>>(Wk, wkb, cc);
  cvt_f32_bf16<<<(cc + 255) / 256, 256, 0, stream>>>(Wv, wvb, cc);
  cvt_f32_bf16<<<(cc + 255) / 256, 256, 0, stream>>>(Wr, wrb, cc);
  cvt_f32_bf16<<<(cc + 255) / 256, 256, 0, stream>>>(Wo, wob, cc);
  cvt_f32_bf16<<<(hc + 255) / 256, 256, 0, stream>>>(Fk, fkb, hc);
  cvt_f32_bf16<<<(hc + 255) / 256, 256, 0, stream>>>(Fv, fvb, hc);
  cvt_f32_bf16<<<(cc + 255) / 256, 256, 0, stream>>>(Fr, frb, cc);

  // ---- attention branch ----
  ln_mix_kernel<<<M, 256, 0, stream>>>(x, ln1_g, ln1_b, amk, amv, amr,
                                       xk, xv, xr, T, C);

  const dim3 gC(C / BN, M / BM);     // (4, 200)
  const dim3 gH(H / BN, M / BM);     // (16, 200)
  gemm_bf16_wmma<0><<<gC, 256, 0, stream>>>(xk, wkb, kb, nullptr, nullptr, nullptr, M, C, C);
  gemm_bf16_wmma<0><<<gC, 256, 0, stream>>>(xv, wvb, vb, nullptr, nullptr, nullptr, M, C, C);
  gemm_bf16_wmma<1><<<gC, 256, 0, stream>>>(xr, wrb, rb, nullptr, nullptr, nullptr, M, C, C);

  __bf16* y = xr;                    // xr dead after the r-GEMM
  wkv_kernel<<<(B * C) / 256, 256, 0, stream>>>(kb, vb, rb, tdec, tfir, y, T, C);

  // x1 = x + (sigmoid(r)*wkv) @ Wo^T  -> d_out
  gemm_bf16_wmma<3><<<gC, 256, 0, stream>>>(y, wob, out, nullptr, x, nullptr, M, C, C);

  // ---- FFN branch ----
  __bf16* fk_a = xk;                 // reuse
  __bf16* fr_a = xv;
  ln_mix_kernel<<<M, 256, 0, stream>>>(out, ln2_g, ln2_b, fmk, fmr, nullptr,
                                       fk_a, fr_a, nullptr, T, C);

  __bf16* kk = (__bf16*)kb;          // spans kb+vb: M*H bf16
  gemm_bf16_wmma<2><<<gH, 256, 0, stream>>>(fk_a, fkb, nullptr, kk, nullptr, nullptr, M, H, C);

  float* kv = rb;                    // reuse
  gemm_bf16_wmma<0><<<gC, 256, 0, stream>>>(kk, fvb, kv, nullptr, nullptr, nullptr, M, C, H);

  // out = x1 + sigmoid(fr @ Fr^T) * kv
  gemm_bf16_wmma<4><<<gC, 256, 0, stream>>>(fr_a, frb, out, nullptr, out, kv, M, C, C);
}
